// GNNEncoder_2611340116103
// MI455X (gfx1250) — matrix operand order repaired and checked by reading, not compiled, not run on hardware
//
#include <hip/hip_runtime.h>

typedef __attribute__((ext_vector_type(16))) _Float16 v16h;
typedef __attribute__((ext_vector_type(8)))  float    v8f;
typedef __attribute__((ext_vector_type(4)))  float    v4f;

#define WMMA_F16(a, b, c) \
  __builtin_amdgcn_wmma_f32_16x16x32_f16(false, (a), false, (b), (short)0, (c), false, false)

// ---------- helpers ----------

// Order-preserving int key for float atomic max (no NaN inputs here).
__device__ __forceinline__ int fkey(float x) {
  int i = __float_as_int(x);
  return (i < 0) ? (i ^ 0x7FFFFFFF) : i;
}
__device__ __forceinline__ float funkey(int i) {
  return __int_as_float((i < 0) ? (i ^ 0x7FFFFFFF) : i);
}

// A fragment for V_WMMA_F32_16X16X32_F16, K-chunk kc (0/1), lane half g.
// Lane g*16+m holds row m; elems 0..15 <-> K = kc*32 + g*16 + e.
__device__ __forceinline__ v16h load_a_frag(const float* __restrict__ hr, int kc, int g) {
  const float* p = hr + kc * 32 + g * 16;
  v4f x0 = *(const v4f*)(p);
  v4f x1 = *(const v4f*)(p + 4);
  v4f y0 = *(const v4f*)(p + 8);
  v4f y1 = *(const v4f*)(p + 12);
  v16h a = {};
#pragma unroll
  for (int t = 0; t < 4; ++t) {
    a[t]      = (_Float16)x0[t];
    a[4 + t]  = (_Float16)x1[t];
    a[8 + t]  = (_Float16)y0[t];
    a[12 + t] = (_Float16)y1[t];
  }
  return a;
}

// ---------- prep: weight transpose->f16, attention folding ----------
// wt[j*64+k] = W[k*64+j]  (so each lane's B fragment is one contiguous 32B run)
__global__ void prep_kernel(const float* __restrict__ w_ne2, const float* __restrict__ w_out,
                            const float* __restrict__ w_gat, const float* __restrict__ att_src,
                            const float* __restrict__ att_dst, const float* __restrict__ w_eg,
                            const float* __restrict__ att_edge, const float* __restrict__ w_ee,
                            const float* __restrict__ b_ee,
                            _Float16* __restrict__ wt_ne2, _Float16* __restrict__ wt_out,
                            _Float16* __restrict__ wt_gat,
                            float* __restrict__ vs, float* __restrict__ vd,
                            float* __restrict__ ce) {
  int t = blockIdx.x * blockDim.x + threadIdx.x;
  if (t < 4096) { int j = t / 64, k = t % 64; wt_ne2[j * 64 + k] = (_Float16)w_ne2[k * 64 + j]; return; }
  t -= 4096;
  if (t < 4096) { int j = t / 64, k = t % 64; wt_out[j * 64 + k] = (_Float16)w_out[k * 64 + j]; return; }
  t -= 4096;
  if (t < 3 * 4096) {
    int l = t / 4096, r = t % 4096, j = r / 64, k = r % 64;
    wt_gat[l * 4096 + j * 64 + k] = (_Float16)w_gat[l * 4096 + k * 64 + j];
    return;
  }
  t -= 3 * 4096;
  if (t < 768) {  // vs[l][k][h] = sum_d w_gat[l][k][h*16+d] * att_src[l][h][d]
    int l = t / 256, r = t % 256, k = r / 4, h = r % 4;
    float s = 0.f;
    for (int d = 0; d < 16; ++d) s += w_gat[l * 4096 + k * 64 + h * 16 + d] * att_src[l * 64 + h * 16 + d];
    vs[l * 256 + k * 4 + h] = s;
    return;
  }
  t -= 768;
  if (t < 768) {
    int l = t / 256, r = t % 256, k = r / 4, h = r % 4;
    float s = 0.f;
    for (int d = 0; d < 16; ++d) s += w_gat[l * 4096 + k * 64 + h * 16 + d] * att_dst[l * 64 + h * 16 + d];
    vd[l * 256 + k * 4 + h] = s;
    return;
  }
  t -= 768;
  if (t < 12) {  // a_e[i,h] = ea[i]*ce1 + ce0  (EIN==1 collapse)
    int l = t / 4, h = t % 4;
    float c0 = 0.f, c1 = 0.f;
    for (int k = 0; k < 64; ++k) {
      float veg = 0.f;
      for (int d = 0; d < 16; ++d) veg += w_eg[l * 4096 + k * 64 + h * 16 + d] * att_edge[l * 64 + h * 16 + d];
      c1 += w_ee[k] * veg;
      c0 += b_ee[k] * veg;
    }
    ce[l * 8 + h] = c0;
    ce[l * 8 + 4 + h] = c1;
  }
}

// ---------- node encoder: h = relu(x@w1+b1) @ w2 + b2 via WMMA (full 16-row tiles) ----------
__global__ void encoder_kernel(const float* __restrict__ x, const float* __restrict__ w1,
                               const float* __restrict__ b1, const _Float16* __restrict__ wt2,
                               const float* __restrict__ b2, float* __restrict__ hout,
                               int ntiles) {
  __shared__ float w1s[6 * 64];
  __shared__ float b1s[64];
  for (int i = threadIdx.x; i < 6 * 64; i += blockDim.x) w1s[i] = w1[i];
  for (int i = threadIdx.x; i < 64; i += blockDim.x) b1s[i] = b1[i];
  __syncthreads();

  int wid = threadIdx.x >> 5;
  int tile = blockIdx.x * (blockDim.x >> 5) + wid;
  if (tile >= ntiles) return;                 // wave-uniform
  int lane = threadIdx.x & 31;
  int g = lane >> 4, m = lane & 15;
  int row = tile * 16 + m;

  float xr[6];
#pragma unroll
  for (int q = 0; q < 6; ++q) xr[q] = x[(size_t)row * 6 + q];

  v16h a0 = {}, a1 = {};
#pragma unroll
  for (int kc = 0; kc < 2; ++kc) {
    v16h a = {};
#pragma unroll
    for (int e = 0; e < 16; ++e) {
      int j = kc * 32 + g * 16 + e;
      float t = b1s[j];
#pragma unroll
      for (int q = 0; q < 6; ++q) t += xr[q] * w1s[q * 64 + j];
      t = (t > 0.f) ? t : 0.f;
      a[e] = (_Float16)t;
    }
    if (kc == 0) a0 = a; else a1 = a;
  }

#pragma unroll
  for (int c = 0; c < 4; ++c) {
    const _Float16* wr = wt2 + (c * 16 + m) * 64;
    v16h bf0 = *(const v16h*)(wr + g * 16);
    v16h bf1 = *(const v16h*)(wr + 32 + g * 16);
    v8f acc = {};
    acc = WMMA_F16(a0, bf0, acc);
    acc = WMMA_F16(a1, bf1, acc);
    int col = c * 16 + m;
    float bb = b2[col];
    float* po = hout + (size_t)(tile * 16 + g * 8) * 64 + col;  // 8 stores, imm offsets
#pragma unroll
    for (int v = 0; v < 8; ++v) po[v * 64] = acc[v] + bb;
  }
}

// ---------- per-layer node GEMM: hw = h @ W (WMMA), plus a_s/a_d = h @ vs/vd ----------
__global__ void gemm64_kernel(const float* __restrict__ hin, const _Float16* __restrict__ wt,
                              const float* __restrict__ bias, float* __restrict__ out,
                              const float* __restrict__ vs, const float* __restrict__ vd,
                              float* __restrict__ as_, float* __restrict__ ad_,
                              int ntiles) {
  int wid = threadIdx.x >> 5;
  int tile = blockIdx.x * (blockDim.x >> 5) + wid;
  if (tile >= ntiles) return;                 // wave-uniform
  int lane = threadIdx.x & 31;
  int g = lane >> 4, m = lane & 15;
  const float* hr = hin + (size_t)(tile * 16 + m) * 64;

  v16h a0 = load_a_frag(hr, 0, g);
  v16h a1 = load_a_frag(hr, 1, g);

#pragma unroll
  for (int c = 0; c < 4; ++c) {
    const _Float16* wr = wt + (c * 16 + m) * 64;
    v16h bf0 = *(const v16h*)(wr + g * 16);       // B: lane half g -> K = g*16 + e
    v16h bf1 = *(const v16h*)(wr + 32 + g * 16);  // kc=1
    v8f acc = {};
    acc = WMMA_F16(a0, bf0, acc);
    acc = WMMA_F16(a1, bf1, acc);
    int col = c * 16 + m;
    float bb = bias ? bias[col] : 0.f;
    float* po = out + (size_t)(tile * 16 + g * 8) * 64 + col;
#pragma unroll
    for (int v = 0; v < 8; ++v) po[v * 64] = acc[v] + bb;
  }

  if (vs != nullptr && g == 0) {  // lanes 0..15: attention logits for this tile's rows
    int r = tile * 16 + m;
    const float* hrow = hin + (size_t)r * 64;
    float s0 = 0, s1 = 0, s2 = 0, s3 = 0, d0 = 0, d1 = 0, d2 = 0, d3 = 0;
    for (int k = 0; k < 64; ++k) {
      float hv = hrow[k];
      const float* a = vs + k * 4;
      const float* b = vd + k * 4;
      s0 += hv * a[0]; s1 += hv * a[1]; s2 += hv * a[2]; s3 += hv * a[3];
      d0 += hv * b[0]; d1 += hv * b[1]; d2 += hv * b[2]; d3 += hv * b[3];
    }
    as_[r * 4 + 0] = s0; as_[r * 4 + 1] = s1; as_[r * 4 + 2] = s2; as_[r * 4 + 3] = s3;
    ad_[r * 4 + 0] = d0; ad_[r * 4 + 1] = d1; ad_[r * 4 + 2] = d2; ad_[r * 4 + 3] = d3;
  }
}

// ---------- scalar tail kernels (only launched when n % 16 != 0; empty for N=100000) ----------
__global__ void encoder_tail_kernel(const float* __restrict__ x, const float* __restrict__ w1,
                                    const float* __restrict__ b1, const _Float16* __restrict__ wt2,
                                    const float* __restrict__ b2, float* __restrict__ hout,
                                    int n, int base) {
  int t = base * 64 + blockIdx.x * blockDim.x + threadIdx.x;
  if (t >= n * 64) return;
  int r = t / 64, col = t % 64;
  float acc = b2[col];
  for (int k = 0; k < 64; ++k) {
    float tv = b1[k];
    for (int q = 0; q < 6; ++q) tv += x[(size_t)r * 6 + q] * w1[q * 64 + k];
    tv = (tv > 0.f) ? tv : 0.f;
    acc += tv * (float)wt2[col * 64 + k];
  }
  hout[t] = acc;
}

__global__ void gemm_tail_kernel(const float* __restrict__ hin, const _Float16* __restrict__ wt,
                                 const float* __restrict__ bias, float* __restrict__ out,
                                 const float* __restrict__ vs, const float* __restrict__ vd,
                                 float* __restrict__ as_, float* __restrict__ ad_,
                                 int n, int base) {
  int t = base * 64 + blockIdx.x * blockDim.x + threadIdx.x;
  if (t >= n * 64) return;
  int r = t / 64, col = t % 64;
  const float* hr = hin + (size_t)r * 64;
  float acc = bias ? bias[col] : 0.f;
  for (int k = 0; k < 64; ++k) acc += hr[k] * (float)wt[col * 64 + k];
  out[t] = acc;
  if (vs != nullptr && col < 4) {
    float s = 0.f, d = 0.f;
    for (int k = 0; k < 64; ++k) { s += hr[k] * vs[k * 4 + col]; d += hr[k] * vd[k * 4 + col]; }
    as_[r * 4 + col] = s;
    ad_[r * 4 + col] = d;
  }
}

// ---------- clear per-layer accumulators ----------
__global__ void clear_kernel(float* __restrict__ agg, float* __restrict__ den,
                             int* __restrict__ smax, int n) {
  int i = blockIdx.x * blockDim.x + threadIdx.x;
  if (i < n * 64) agg[i] = 0.f;
  if (i < n * 4) { den[i] = 0.f; smax[i] = (int)0x80000000; }
}

// ---------- edge pass 1: segment max of leaky-relu scores ----------
__global__ void edge_max_kernel(const long long* __restrict__ ei, const float* __restrict__ ea,
                                const float* __restrict__ as_, const float* __restrict__ ad_,
                                const float* __restrict__ ce, int* __restrict__ smax, int e) {
  int i = blockIdx.x * blockDim.x + threadIdx.x;
  if (i >= e) return;
  int s = (int)ei[i], d = (int)ei[(size_t)e + i];
  float w = ea[i];
#pragma unroll
  for (int h = 0; h < 4; ++h) {
    float sc = as_[s * 4 + h] + ad_[d * 4 + h] + w * ce[4 + h] + ce[h];
    sc = (sc > 0.f) ? sc : 0.2f * sc;
    atomicMax(&smax[d * 4 + h], fkey(sc));
  }
}

// ---------- edge pass 2: softmax denominator ----------
__global__ void edge_den_kernel(const long long* __restrict__ ei, const float* __restrict__ ea,
                                const float* __restrict__ as_, const float* __restrict__ ad_,
                                const float* __restrict__ ce, const int* __restrict__ smax,
                                float* __restrict__ den, int e) {
  int i = blockIdx.x * blockDim.x + threadIdx.x;
  if (i >= e) return;
  int s = (int)ei[i], d = (int)ei[(size_t)e + i];
  float w = ea[i];
#pragma unroll
  for (int h = 0; h < 4; ++h) {
    float sc = as_[s * 4 + h] + ad_[d * 4 + h] + w * ce[4 + h] + ce[h];
    sc = (sc > 0.f) ? sc : 0.2f * sc;
    float ex = __expf(sc - funkey(smax[d * 4 + h]));
    atomicAdd(&den[d * 4 + h], ex);
  }
}

// ---------- edge pass 3: alpha-weighted message scatter ----------
__global__ void edge_scatter_kernel(const long long* __restrict__ ei, const float* __restrict__ ea,
                                    const float* __restrict__ as_, const float* __restrict__ ad_,
                                    const float* __restrict__ ce, const int* __restrict__ smax,
                                    const float* __restrict__ den, const float* __restrict__ hw,
                                    float* __restrict__ agg, int e) {
  int i = blockIdx.x * blockDim.x + threadIdx.x;
  if (i >= e) return;
  int s = (int)ei[i], d = (int)ei[(size_t)e + i];
  float w = ea[i];
  float alpha[4];
#pragma unroll
  for (int h = 0; h < 4; ++h) {
    float sc = as_[s * 4 + h] + ad_[d * 4 + h] + w * ce[4 + h] + ce[h];
    sc = (sc > 0.f) ? sc : 0.2f * sc;
    float ex = __expf(sc - funkey(smax[d * 4 + h]));
    alpha[h] = ex / (den[d * 4 + h] + 1e-16f);
  }
  const float* hs = hw + (size_t)s * 64;
  float* ag = agg + (size_t)d * 64;
#pragma unroll
  for (int j = 0; j < 64; j += 4) {
    v4f v = *(const v4f*)(hs + j);
    float a = alpha[j >> 4];
    atomicAdd(ag + j + 0, a * v[0]);
    atomicAdd(ag + j + 1, a * v[1]);
    atomicAdd(ag + j + 2, a * v[2]);
    atomicAdd(ag + j + 3, a * v[3]);
  }
}

// ---------- node update: h = LN(h + elu(agg + b_gat)) ; wave per node ----------
__global__ void update_kernel(float* __restrict__ h, const float* __restrict__ agg,
                              const float* __restrict__ bg, const float* __restrict__ lg,
                              const float* __restrict__ lb, int n) {
  int gid = blockIdx.x * blockDim.x + threadIdx.x;
  int node = gid >> 5;
  if (node >= n) return;                       // wave-uniform
  int lane = threadIdx.x & 31;
  int j0 = lane * 2, j1 = j0 + 1;
  const float* ag = agg + (size_t)node * 64;
  float* hr = h + (size_t)node * 64;

  float g0 = ag[j0] + bg[j0];
  float g1 = ag[j1] + bg[j1];
  float u0 = hr[j0] + ((g0 > 0.f) ? g0 : (__expf(g0) - 1.f));
  float u1 = hr[j1] + ((g1 > 0.f) ? g1 : (__expf(g1) - 1.f));

  float sum = u0 + u1;
#pragma unroll
  for (int off = 16; off; off >>= 1) sum += __shfl_xor(sum, off, 32);
  float mean = sum * (1.f / 64.f);
  float e0 = u0 - mean, e1 = u1 - mean;
  float vsum = e0 * e0 + e1 * e1;
#pragma unroll
  for (int off = 16; off; off >>= 1) vsum += __shfl_xor(vsum, off, 32);
  float rs = rsqrtf(vsum * (1.f / 64.f) + 1e-5f);
  hr[j0] = e0 * rs * lg[j0] + lb[j0];
  hr[j1] = e1 * rs * lg[j1] + lb[j1];
}

// ---------- host ----------
extern "C" void kernel_launch(void* const* d_in, const int* in_sizes, int n_in,
                              void* d_out, int out_size, void* d_ws, size_t ws_size,
                              hipStream_t stream) {
  const float* x        = (const float*)d_in[0];
  const long long* eidx = (const long long*)d_in[1];  // jnp.int64 per reference
  const float* eattr    = (const float*)d_in[2];
  const float* w_ne1    = (const float*)d_in[3];
  const float* b_ne1    = (const float*)d_in[4];
  const float* w_ne2    = (const float*)d_in[5];
  const float* b_ne2    = (const float*)d_in[6];
  const float* w_ee     = (const float*)d_in[7];
  const float* b_ee     = (const float*)d_in[8];
  const float* w_gat    = (const float*)d_in[9];
  const float* w_eg     = (const float*)d_in[10];
  const float* att_src  = (const float*)d_in[11];
  const float* att_dst  = (const float*)d_in[12];
  const float* att_edge = (const float*)d_in[13];
  const float* b_gat    = (const float*)d_in[14];
  const float* ln_g     = (const float*)d_in[15];
  const float* ln_b     = (const float*)d_in[16];
  const float* w_out    = (const float*)d_in[17];
  const float* b_out    = (const float*)d_in[18];
  float* out = (float*)d_out;

  const int n = in_sizes[0] / 6;   // nodes
  const int e = in_sizes[2];       // edges (edge_attr is [E,1])

  // bump-allocate workspace (256B aligned)
  char* base = (char*)d_ws;
  size_t off = 0;
  auto bump = [&](size_t bytes) -> void* {
    void* p = base + off;
    off = (off + bytes + 255) & ~(size_t)255;
    return p;
  };
  float*     h      = (float*)bump((size_t)n * 64 * 4);
  float*     hw     = (float*)bump((size_t)n * 64 * 4);
  float*     agg    = (float*)bump((size_t)n * 64 * 4);
  float*     a_s    = (float*)bump((size_t)n * 4 * 4);
  float*     a_d    = (float*)bump((size_t)n * 4 * 4);
  int*       smax   = (int*)bump((size_t)n * 4 * 4);
  float*     den    = (float*)bump((size_t)n * 4 * 4);
  _Float16*  wt_ne2 = (_Float16*)bump(4096 * 2);
  _Float16*  wt_out = (_Float16*)bump(4096 * 2);
  _Float16*  wt_gat = (_Float16*)bump(3 * 4096 * 2);
  float*     vs     = (float*)bump(3 * 256 * 4);
  float*     vd     = (float*)bump(3 * 256 * 4);
  float*     ce     = (float*)bump(3 * 8 * 4);

  const int ntiles  = n / 16;                    // complete tiles (WMMA path)
  const int rem     = n - ntiles * 16;           // ragged tail rows (0 for N=100000)
  const int tblocks = (rem * 64 + 255) / 256;
  const int gblocks = (ntiles + 7) / 8;          // 8 waves (tiles) per 256-thread block
  const int eblocks = (e + 255) / 256;
  const int cblocks = (n * 64 + 255) / 256;
  const int ublocks = (n + 7) / 8;               // wave per node, 8 per block

  prep_kernel<<<(5 * 4096 + 2 * 768 + 12 + 255) / 256, 256, 0, stream>>>(
      w_ne2, w_out, w_gat, att_src, att_dst, w_eg, att_edge, w_ee, b_ee,
      wt_ne2, wt_out, wt_gat, vs, vd, ce);

  encoder_kernel<<<gblocks, 256, 0, stream>>>(x, w_ne1, b_ne1, wt_ne2, b_ne2, h, ntiles);
  if (rem)
    encoder_tail_kernel<<<tblocks, 256, 0, stream>>>(x, w_ne1, b_ne1, wt_ne2, b_ne2, h,
                                                     n, ntiles * 16);

  for (int l = 0; l < 3; ++l) {
    clear_kernel<<<cblocks, 256, 0, stream>>>(agg, den, smax, n);
    gemm64_kernel<<<gblocks, 256, 0, stream>>>(h, wt_gat + (size_t)l * 4096, nullptr, hw,
                                               vs + l * 256, vd + l * 256, a_s, a_d, ntiles);
    if (rem)
      gemm_tail_kernel<<<tblocks, 256, 0, stream>>>(h, wt_gat + (size_t)l * 4096, nullptr, hw,
                                                    vs + l * 256, vd + l * 256, a_s, a_d,
                                                    n, ntiles * 16);
    edge_max_kernel<<<eblocks, 256, 0, stream>>>(eidx, eattr, a_s, a_d, ce + l * 8, smax, e);
    edge_den_kernel<<<eblocks, 256, 0, stream>>>(eidx, eattr, a_s, a_d, ce + l * 8, smax, den, e);
    edge_scatter_kernel<<<eblocks, 256, 0, stream>>>(eidx, eattr, a_s, a_d, ce + l * 8, smax,
                                                     den, hw, agg, e);
    update_kernel<<<ublocks, 256, 0, stream>>>(h, agg, b_gat + l * 64,
                                               ln_g + l * 64, ln_b + l * 64, n);
  }

  gemm64_kernel<<<gblocks, 256, 0, stream>>>(h, wt_out, b_out, out,
                                             nullptr, nullptr, nullptr, nullptr, ntiles);
  if (rem)
    gemm_tail_kernel<<<tblocks, 256, 0, stream>>>(h, wt_out, b_out, out,
                                                  nullptr, nullptr, nullptr, nullptr,
                                                  n, ntiles * 16);
}
